// PairPot_24034636989173
// MI455X (gfx1250) — matrix operand order repaired and checked by reading, not compiled, not run on hardware
//
#include <hip/hip_runtime.h>

typedef __attribute__((ext_vector_type(16))) _Float16 v16h;
typedef __attribute__((ext_vector_type(8)))  float    v8f;

#define CUTOFF2 6.25f

__global__ __launch_bounds__(256) void PairPot_kernel(
    const float* __restrict__ xyz, const float* __restrict__ cell,
    const float* __restrict__ w1,  const float* __restrict__ b1,
    const float* __restrict__ w2,  const float* __restrict__ b2,
    float* __restrict__ out, int n)
{
    const int lid  = threadIdx.x & 31;   // lane in wave32
    const int wid  = threadIdx.x >> 5;   // wave in block (0..7)
    const int i    = blockIdx.x;         // pair-matrix row

    const float bx = cell[0], by = cell[1], bz = cell[2];
    const float xi = xyz[3*i+0], yi = xyz[3*i+1], zi = xyz[3*i+2];
    const float b2s = b2[0];

    // ---- Per-lane layer-1 weights in WMMA A-operand order (16-bit A 16x32):
    //   lanes 0-15 : VGPR v holds K = {2v,2v+1}        (v<4)  / {16+2(v-4), ...} (v>=4)
    //   lanes 16-31: VGPR v holds K = {8+2v,8+2v+1}    (v<4)  / {24+2(v-4), ...} (v>=4)
    // ---- B operand (32x16): w2 replicated across all 16 columns so that
    //   D[m][n] == e_m for every n; per-lane K striping matches A: lanes 0-15 hold
    //   K=0..7,16..23 (packed 2/VGPR), lanes 16-31 hold K=8..15,24..31.
    _Float16 w1h[32], b1h[32];
    v16h bmat0, bmat1;
    const int half = (lid < 16) ? 0 : 1;
    #pragma unroll
    for (int t = 0; t < 2; ++t) {
        #pragma unroll
        for (int v = 0; v < 8; ++v) {
            int kA = t*32 + ((v < 4) ? (half ? 8 : 0)  + 2*v
                                     : (half ? 24 : 16) + 2*(v-4));
            w1h[t*16 + 2*v+0] = (_Float16)w1[kA+0];
            w1h[t*16 + 2*v+1] = (_Float16)w1[kA+1];
            b1h[t*16 + 2*v+0] = (_Float16)b1[kA+0];
            b1h[t*16 + 2*v+1] = (_Float16)b1[kA+1];
            int kB = kA;
            if (t == 0) { bmat0[2*v+0] = (_Float16)w2[kB+0];
                          bmat0[2*v+1] = (_Float16)w2[kB+1]; }
            else        { bmat1[2*v+0] = (_Float16)w2[kB+0];
                          bmat1[2*v+1] = (_Float16)w2[kB+1]; }
        }
    }

    const int  m      = lid & 15;                       // pair this lane computes
    const int  msel   = (lid & 7) | ((lid & 16) >> 1);  // pair this lane stores
    const bool storer = (lid & 8) == 0;                 // 16 store lanes -> 64B run

    for (int jb = wid * 16; jb < n; jb += 8 * 16) {
        // ---- distance + PBC + mask (clamped index: no EXEC divergence) ----
        int j  = jb + m;
        int jc = (j < n) ? j : 0;
        float dx = xyz[3*jc+0] - xi;
        float dy = xyz[3*jc+1] - yi;
        float dz = xyz[3*jc+2] - zi;
        dx += ((dx < -0.5f*bx) ? bx : 0.f) - ((dx >= 0.5f*bx) ? bx : 0.f);
        dy += ((dy < -0.5f*by) ? by : 0.f) - ((dy >= 0.5f*by) ? by : 0.f);
        dz += ((dz < -0.5f*bz) ? bz : 0.f) - ((dz >= 0.5f*bz) ? bz : 0.f);
        float r2    = dx*dx + dy*dy + dz*dz;
        float maskf = (r2 < CUTOFF2 && r2 != 0.f) ? 1.f : 0.f;
        _Float16 rh = (_Float16)sqrtf(r2);

        // ---- layer 1 in f16, built straight into A-operand registers ----
        v16h a0, a1;
        #pragma unroll
        for (int e = 0; e < 16; ++e) {
            _Float16 h0 = rh * w1h[e]      + b1h[e];
            _Float16 h1 = rh * w1h[16 + e] + b1h[16 + e];
            a0[e] = (h0 > (_Float16)0) ? h0 : (_Float16)0;
            a1[e] = (h1 > (_Float16)0) ? h1 : (_Float16)0;
        }

        // ---- layer 2 on the matrix pipe: D[m][*] = e_m (f32 accumulate) ----
        v8f c = {};
        c = __builtin_amdgcn_wmma_f32_16x16x32_f16(false, a0, false, bmat0,
                                                   (short)0, c, false, false);
        c = __builtin_amdgcn_wmma_f32_16x16x32_f16(false, a1, false, bmat1,
                                                   (short)0, c, false, false);

        // ---- fetch mask for the pair this lane stores (one lds bpermute) ----
        float maskSel = __int_as_float(
            __builtin_amdgcn_ds_bpermute(msel * 4, __float_as_int(maskf)));

        // ---- select e_{msel}: row M = (lid&7) + 8*(lid>=16) lives in c[lid&7]
        float e01 = (lid & 1) ? c[1] : c[0];
        float e23 = (lid & 1) ? c[3] : c[2];
        float e45 = (lid & 1) ? c[5] : c[4];
        float e67 = (lid & 1) ? c[7] : c[6];
        float eA  = (lid & 2) ? e23 : e01;
        float eB  = (lid & 2) ? e67 : e45;
        float ev  = (lid & 4) ? eB  : eA;

        if (storer && (jb + msel) < n)
            out[(size_t)i * n + jb + msel] = (ev + b2s) * maskSel;
    }
}

extern "C" void kernel_launch(void* const* d_in, const int* in_sizes, int n_in,
                              void* d_out, int out_size, void* d_ws, size_t ws_size,
                              hipStream_t stream) {
    const float* xyz  = (const float*)d_in[0];
    const float* cell = (const float*)d_in[1];
    const float* w1   = (const float*)d_in[2];
    const float* b1   = (const float*)d_in[3];
    const float* w2   = (const float*)d_in[4];
    const float* b2   = (const float*)d_in[5];
    float* out = (float*)d_out;
    const int n = in_sizes[0] / 3;   // N = 2048
    PairPot_kernel<<<n, 256, 0, stream>>>(xyz, cell, w1, b1, w2, b2, out, n);
}
